// MambaLayer_30150670418085
// MI455X (gfx1250) — compile-verified, matrix operand order checked by reading, not run on hardware
//
#include <hip/hip_runtime.h>
#include <hip/hip_bf16.h>
#include <stdint.h>

// ---------------- problem constants ----------------
#define NUM_LAYERS 2
#define D_MODEL    256
#define D_STATE    16
#define D_CONV     4
#define D_INNER    512            // D_MODEL * 2
#define DT_RANK    16
#define BATCH      8
#define SEQ        2048
#define MTOK       (BATCH * SEQ)  // 16384 tokens
#define EPSF       1e-5f
#define XDBL_LD    64             // x_dbl padded stride (48 -> 64)

// GEMM tiling
#define BM   128                  // 8 waves x 16 rows
#define BN   64                   // 4 wmma tiles wide
#define BK   64                   // K per LDS stage
#define PADH 8
#define ASTR (BK + PADH)          // 72 halves per LDS row (144 B, bank-spread)

typedef __attribute__((ext_vector_type(8)))  float  v8f;
typedef __attribute__((ext_vector_type(16))) __bf16 v16bf;

union Frag { v16bf v; uint4 q[2]; };

// round-to-nearest-even f32 -> bf16 (bit pattern)
__device__ __forceinline__ uint16_t f2bf(float x) {
    uint32_t u = __float_as_uint(x);
    uint32_t r = (u + 0x7FFFu + ((u >> 16) & 1u)) >> 16;
    return (uint16_t)r;
}
__device__ __forceinline__ float siluf(float x) {
    return x / (1.f + __expf(-x));
}
// CDNA5 async DMA: global -> LDS, 16 bytes, tracked by ASYNCcnt
__device__ __forceinline__ void async_b128(uint32_t lds_off, const void* gptr) {
    asm volatile("global_load_async_to_lds_b128 %0, %1, off"
                 :: "v"(lds_off), "v"(gptr) : "memory");
}
__device__ __forceinline__ uint32_t lds_off32(const void* p) {
    return (uint32_t)(uintptr_t)p;   // low 32 bits of flat shared addr = LDS offset
}

// ---------------- f32 -> bf16 convert ----------------
__global__ void cvt_bf16_kernel(const float* __restrict__ src,
                                uint16_t* __restrict__ dst, int n) {
    int i = blockIdx.x * blockDim.x + threadIdx.x;
    if (i < n) dst[i] = f2bf(src[i]);
}
// convert with zero row padding: dst has rows_dst rows of K, src rows_src rows
__global__ void cvt_pad_kernel(const float* __restrict__ src,
                               uint16_t* __restrict__ dst,
                               int rows_src, int K, int n) {
    int i = blockIdx.x * blockDim.x + threadIdx.x;
    if (i < n) {
        int r = i / K;
        dst[i] = (r < rows_src) ? f2bf(src[i]) : (uint16_t)0;
    }
}

// ---------------- LayerNorm: one wave32 per token ----------------
__global__ void ln_kernel(const float* __restrict__ x,
                          const float* __restrict__ g,
                          const float* __restrict__ b,
                          uint16_t* __restrict__ h16) {
    int lane = threadIdx.x & 31;
    int wv   = threadIdx.x >> 5;
    int tok  = blockIdx.x * 8 + wv;
    const float* row = x + (size_t)tok * D_MODEL;

    float v[8];
    float s = 0.f;
#pragma unroll
    for (int r = 0; r < 8; ++r) { v[r] = row[lane + 32 * r]; s += v[r]; }
#pragma unroll
    for (int off = 16; off; off >>= 1) s += __shfl_xor(s, off, 32);
    float mean = s * (1.f / D_MODEL);

    float q = 0.f;
#pragma unroll
    for (int r = 0; r < 8; ++r) { float d = v[r] - mean; q += d * d; }
#pragma unroll
    for (int off = 16; off; off >>= 1) q += __shfl_xor(q, off, 32);
    float rstd = rsqrtf(q * (1.f / D_MODEL) + EPSF);

#pragma unroll
    for (int r = 0; r < 8; ++r) {
        int idx = lane + 32 * r;
        float o = (v[r] - mean) * rstd * g[idx] + b[idx];
        h16[(size_t)tok * D_MODEL + idx] = f2bf(o);
    }
}

// ---------------- WMMA GEMM with async-LDS double buffering ----------------
// C[M,N] = A[M,K] * W[N,K]^T ; A, W bf16 row-major (K contiguous).
// Block = 256 threads (8 waves), tile BM=128 x BN=64, K staged in BK=64 chunks.
// mode 0: C = acc     mode 1: C = res + ls[col] * acc
__global__ __launch_bounds__(256)
void gemm_wmma_kernel(const uint16_t* __restrict__ A,
                      const uint16_t* __restrict__ W,
                      int N, int K,
                      float* __restrict__ C, int mode,
                      const float* __restrict__ res,
                      const float* __restrict__ ls) {
    __shared__ uint16_t shA[2][BM * ASTR];
    __shared__ uint16_t shB[2][BN * ASTR];

    const int t    = threadIdx.x;
    const int lane = t & 31;
    const int wv   = t >> 5;
    const int m0   = blockIdx.x * BM;
    const int n0   = blockIdx.y * BN;
    const int half = lane >> 4;   // K-group per ISA bf16 fragment layout
    const int lr   = lane & 15;   // row (A) / col (B) within 16x16 tile

    // cooperative copy pattern: 256 threads x 16B, 32 rows x 128B per pass
    const int crow = t >> 3;        // 0..31
    const int ccol = (t & 7) * 8;   // halves

    v8f acc[4];
#pragma unroll
    for (int j = 0; j < 4; ++j)
#pragma unroll
        for (int r = 0; r < 8; ++r) acc[j][r] = 0.f;

    auto issue = [&](int buf, int kb) {
#pragma unroll
        for (int i = 0; i < 4; ++i) {           // A: 128 rows
            int row = i * 32 + crow;
            async_b128(lds_off32(&shA[buf][row * ASTR + ccol]),
                       A + (size_t)(m0 + row) * K + kb + ccol);
        }
#pragma unroll
        for (int i = 0; i < 2; ++i) {           // B: 64 rows (= out cols)
            int row = i * 32 + crow;
            async_b128(lds_off32(&shB[buf][row * ASTR + ccol]),
                       W + (size_t)(n0 + row) * K + kb + ccol);
        }
    };

    issue(0, 0);
    int it = 0;
    for (int kb = 0; kb < K; kb += BK, ++it) {
        const int cur  = it & 1;
        const bool more = (kb + BK) < K;
        if (more) {
            issue(cur ^ 1, kb + BK);
            asm volatile("s_wait_asynccnt 0x6" ::: "memory"); // prev group done
        } else {
            asm volatile("s_wait_asynccnt 0x0" ::: "memory");
        }
        __syncthreads();

#pragma unroll
        for (int k0 = 0; k0 < BK; k0 += 32) {
            Frag fa;
            const uint16_t* pa = &shA[cur][(wv * 16 + lr) * ASTR + k0 + half * 8];
            fa.q[0] = *reinterpret_cast<const uint4*>(pa);
            fa.q[1] = *reinterpret_cast<const uint4*>(pa + 16);
#pragma unroll
            for (int j = 0; j < 4; ++j) {
                Frag fb;
                const uint16_t* pb =
                    &shB[cur][(j * 16 + lr) * ASTR + k0 + half * 8];
                fb.q[0] = *reinterpret_cast<const uint4*>(pb);
                fb.q[1] = *reinterpret_cast<const uint4*>(pb + 16);
                acc[j] = __builtin_amdgcn_wmma_f32_16x16x32_bf16(
                    false, fa.v, false, fb.v, (short)0, acc[j], false, false);
            }
        }
        __syncthreads();   // protect buffer reuse: next iteration refills it
    }

    // C/D layout: lanes 0-15 -> rows +0..7, lanes 16-31 -> rows +8..15
    const int rowbase = m0 + wv * 16 + 8 * half;
#pragma unroll
    for (int j = 0; j < 4; ++j) {
        int col = n0 + 16 * j + lr;
#pragma unroll
        for (int r = 0; r < 8; ++r) {
            size_t idx = (size_t)(rowbase + r) * N + col;
            float v = acc[j][r];
            if (mode == 1) v = res[idx] + ls[col] * v;
            C[idx] = v;
        }
    }
}

// ---------------- depthwise causal conv (width 4) + bias + SiLU ----------------
__global__ void conv_silu_kernel(const float* __restrict__ xz,
                                 const float* __restrict__ cw,
                                 const float* __restrict__ cb,
                                 float* __restrict__ u,
                                 uint16_t* __restrict__ u16) {
    size_t idx = (size_t)blockIdx.x * blockDim.x + threadIdx.x; // < MTOK*512
    int d        = (int)(idx & (D_INNER - 1));
    size_t tok   = idx >> 9;
    int l        = (int)(tok & (SEQ - 1));
    size_t bbase = tok - l;            // b*SEQ

    float s = cb[d];
#pragma unroll
    for (int j = 0; j < D_CONV; ++j) {
        int lj = l - (D_CONV - 1) + j;
        if (lj >= 0)
            s += cw[d * D_CONV + j] *
                 xz[((size_t)(bbase + lj)) * (2 * D_INNER) + d];
    }
    s = siluf(s);
    u[idx]   = s;
    u16[idx] = f2bf(s);
}

// ---------------- selective scan: block = batch, thread = channel ----------------
// 16 timesteps of x_dbl rows staged per LDS chunk -> 2 barriers / 16 steps
#define TCHUNK 16
__global__ __launch_bounds__(512)
void scan_kernel(const float* __restrict__ xdbl,  // [MTOK, XDBL_LD]
                 const float* __restrict__ u,     // [MTOK,512]
                 const float* __restrict__ xz,    // z at col 512+
                 const float* __restrict__ dtw,   // [512,16]
                 const float* __restrict__ dtb,   // [512]
                 const float* __restrict__ alog,  // [512,16]
                 const float* __restrict__ Dp,    // [512]
                 uint16_t* __restrict__ y16) {    // [MTOK,512]
    __shared__ float rows[TCHUNK][XDBL_LD];
    const int d = threadIdx.x;
    const int b = blockIdx.x;

    float wdt[DT_RANK], Ad[D_STATE], h[D_STATE];
#pragma unroll
    for (int r = 0; r < DT_RANK; ++r) wdt[r] = dtw[d * DT_RANK + r];
#pragma unroll
    for (int n = 0; n < D_STATE; ++n) {
        Ad[n] = -__expf(alog[d * D_STATE + n]);
        h[n]  = 0.f;
    }
    const float bias = dtb[d];
    const float Dd   = Dp[d];

    for (int t0 = 0; t0 < SEQ; t0 += TCHUNK) {
        const size_t tok0 = (size_t)b * SEQ + t0;
        // stage TCHUNK x 64 floats cooperatively (1024 elems / 512 thr)
#pragma unroll
        for (int e = 0; e < 2; ++e) {
            int ix = e * 512 + d;
            rows[ix >> 6][ix & 63] = xdbl[(tok0 + (ix >> 6)) * XDBL_LD + (ix & 63)];
        }
        __syncthreads();

        for (int tt = 0; tt < TCHUNK; ++tt) {
            const size_t tok = tok0 + tt;
            float dtv = bias;
#pragma unroll
            for (int r = 0; r < DT_RANK; ++r) dtv += rows[tt][r] * wdt[r];
            dtv = (dtv > 20.f) ? dtv : log1pf(__expf(dtv));  // softplus

            float ut = u[tok * D_INNER + d];
            float du = dtv * ut;
            float y  = 0.f;
#pragma unroll
            for (int n = 0; n < D_STATE; ++n) {
                float dA = __expf(dtv * Ad[n]);
                h[n] = dA * h[n] + du * rows[tt][DT_RANK + n];
                y += h[n] * rows[tt][DT_RANK + D_STATE + n];
            }
            y += ut * Dd;
            float zv = xz[tok * (2 * D_INNER) + D_INNER + d];
            y *= siluf(zv);
            y16[tok * D_INNER + d] = f2bf(y);
        }
        __syncthreads();
    }
}

// ---------------- host launch ----------------
extern "C" void kernel_launch(void* const* d_in, const int* in_sizes, int n_in,
                              void* d_out, int out_size, void* d_ws, size_t ws_size,
                              hipStream_t stream) {
    (void)in_sizes; (void)n_in; (void)out_size; (void)ws_size;
    const float* x_in      = (const float*)d_in[0];
    const float* ln_g      = (const float*)d_in[1];
    const float* ln_b      = (const float*)d_in[2];
    const float* in_proj_w = (const float*)d_in[3];
    const float* conv_w    = (const float*)d_in[4];
    const float* conv_b    = (const float*)d_in[5];
    const float* x_proj_w  = (const float*)d_in[6];
    const float* dt_proj_w = (const float*)d_in[7];
    const float* dt_proj_b = (const float*)d_in[8];
    const float* A_log     = (const float*)d_in[9];
    const float* Dp        = (const float*)d_in[10];
    const float* out_proj_w= (const float*)d_in[11];
    const float* lscale    = (const float*)d_in[12];

    char* ws = (char*)d_ws;
    size_t off = 0;
    auto take = [&](size_t bytes) { char* p = ws + off; off += (bytes + 255) & ~(size_t)255; return p; };

    float*    xz     = (float*)   take((size_t)MTOK * 2 * D_INNER * 4); // 64 MB
    float*    ubuf   = (float*)   take((size_t)MTOK * D_INNER * 4);     // 32 MB
    uint16_t* h16    = (uint16_t*)take((size_t)MTOK * D_MODEL * 2);     //  8 MB
    uint16_t* u16    = (uint16_t*)take((size_t)MTOK * D_INNER * 2);     // 16 MB
    uint16_t* y16    = (uint16_t*)take((size_t)MTOK * D_INNER * 2);     // 16 MB
    float*    xdbl   = (float*)   take((size_t)MTOK * XDBL_LD * 4);     //  4 MB
    float*    xbuf   = (float*)   take((size_t)MTOK * D_MODEL * 4);     // 16 MB
    uint16_t* w_in16 = (uint16_t*)take((size_t)2 * D_INNER * D_MODEL * 2);
    uint16_t* w_xp16 = (uint16_t*)take((size_t)XDBL_LD * D_INNER * 2);  // padded 64 rows
    uint16_t* w_ot16 = (uint16_t*)take((size_t)D_MODEL * D_INNER * 2);

    for (int i = 0; i < NUM_LAYERS; ++i) {
        const float* xin  = (i == 0) ? x_in : xbuf;
        float*       xout = (i == 0) ? xbuf : (float*)d_out;

        const float* Wi  = in_proj_w + (size_t)i * 2 * D_INNER * D_MODEL;
        const float* Wxp = x_proj_w  + (size_t)i * 48 * D_INNER;
        const float* Wot = out_proj_w+ (size_t)i * D_MODEL * D_INNER;

        // weight conversion to bf16 (x_proj padded 48 -> 64 rows)
        {
            int n1 = 2 * D_INNER * D_MODEL;
            cvt_bf16_kernel<<<(n1 + 255) / 256, 256, 0, stream>>>(Wi, w_in16, n1);
            int n2 = XDBL_LD * D_INNER;
            cvt_pad_kernel<<<(n2 + 255) / 256, 256, 0, stream>>>(Wxp, w_xp16, 48, D_INNER, n2);
            int n3 = D_MODEL * D_INNER;
            cvt_bf16_kernel<<<(n3 + 255) / 256, 256, 0, stream>>>(Wot, w_ot16, n3);
        }

        // layernorm -> bf16 activations
        ln_kernel<<<MTOK / 8, 256, 0, stream>>>(
            xin, ln_g + (size_t)i * D_MODEL, ln_b + (size_t)i * D_MODEL, h16);

        // in_proj: [16384,256] x [256,1024] -> xz
        gemm_wmma_kernel<<<dim3(MTOK / BM, (2 * D_INNER) / BN), 256, 0, stream>>>(
            h16, w_in16, 2 * D_INNER, D_MODEL, xz, 0, nullptr, nullptr);

        // depthwise conv + SiLU
        conv_silu_kernel<<<(MTOK * D_INNER) / 256, 256, 0, stream>>>(
            xz, conv_w + (size_t)i * D_INNER * D_CONV,
            conv_b + (size_t)i * D_INNER, ubuf, u16);

        // x_proj: [16384,512] x [512,64(padded 48)] -> xdbl
        gemm_wmma_kernel<<<dim3(MTOK / BM, XDBL_LD / BN), 256, 0, stream>>>(
            u16, w_xp16, XDBL_LD, D_INNER, xdbl, 0, nullptr, nullptr);

        // selective scan (dt-softplus fused), gated by SiLU(z) -> y16
        scan_kernel<<<BATCH, D_INNER, 0, stream>>>(
            xdbl, ubuf, xz,
            dt_proj_w + (size_t)i * D_INNER * DT_RANK,
            dt_proj_b + (size_t)i * D_INNER,
            A_log + (size_t)i * D_INNER * D_STATE,
            Dp + (size_t)i * D_INNER, y16);

        // out_proj + fused residual + layer_scale
        gemm_wmma_kernel<<<dim3(MTOK / BM, D_MODEL / BN), 256, 0, stream>>>(
            y16, w_ot16, D_MODEL, D_INNER, xout, 1,
            xin, lscale + (size_t)i * D_MODEL);
    }
}
